// Wrapper_56899726738120
// MI455X (gfx1250) — matrix operand order repaired and checked by reading, not compiled, not run on hardware
//
#include <hip/hip_runtime.h>
#include <hip/hip_fp16.h>

typedef __attribute__((ext_vector_type(16))) _Float16 v16h;
typedef __attribute__((ext_vector_type(8)))  _Float16 v8h;
typedef __attribute__((ext_vector_type(8)))  float    v8f;
typedef __attribute__((ext_vector_type(4)))  float    v4f;

#define BN   65536
#define DIM  512
#define NE   8
#define NEP  16          // decision cols padded 8 -> 16 for WMMA
#define MAXT (BN/16 + 16)

// ---------- fragment helpers (layouts per CDNA5 ISA 7.12.2) ----------
// A 16x32 f16: lane(0..15)->M=lane, K=0..7 & 16..23 ; lane(16..31)->M=lane-16, K=8..15 & 24..31
__device__ __forceinline__ v16h load_a_f16(const _Float16* p /* = &row[kstep + khalf*8] */) {
    union { v16h v; v8h h[2]; } u;
    u.h[0] = *(const v8h*)(p);
    u.h[1] = *(const v8h*)(p + 16);
    return u.v;
}
__device__ __forceinline__ v16h load_a_f32(const float* p) {
    v4f a0 = *(const v4f*)(p);      v4f a1 = *(const v4f*)(p + 4);
    v4f a2 = *(const v4f*)(p + 16); v4f a3 = *(const v4f*)(p + 20);
    v16h r;
#pragma unroll
    for (int i = 0; i < 4; ++i) {
        r[i]      = (_Float16)a0[i];
        r[4 + i]  = (_Float16)a1[i];
        r[8 + i]  = (_Float16)a2[i];
        r[12 + i] = (_Float16)a3[i];
    }
    return r;
}
// B 32x16 f16: lane(0..15)->N=lane, K=0..7 & 16..23 ; lane(16..31)->N=lane-16, K=8..15 & 24..31
__device__ __forceinline__ v16h load_b(const _Float16* p /* = &W[(kstep+khalf*8)*ldn + n0 + l16] */, int ld) {
    v16h r;
#pragma unroll
    for (int s = 0; s < 8; ++s) {
        r[s]     = p[(size_t)s * ld];
        r[8 + s] = p[(size_t)(16 + s) * ld];
    }
    return r;
}
#define WMMA(acc, a, b) \
    __builtin_amdgcn_wmma_f32_16x16x32_f16(false, (a), false, (b), (short)0, (acc), false, false)

// ---------- K0: weight conversion fp32 -> f16 (+pad decision cols), zero counters ----------
__global__ void k_prep(const float* __restrict__ Wr, const float* __restrict__ Wd,
                       const float* __restrict__ Wl, const float* __restrict__ Wld,
                       _Float16* __restrict__ Wr16, _Float16* __restrict__ Wd16,
                       _Float16* __restrict__ Wl16, _Float16* __restrict__ Wld16,
                       int* __restrict__ counts, int* __restrict__ cursors) {
    int i = blockIdx.x * blockDim.x + threadIdx.x;
    int stride = gridDim.x * blockDim.x;
    if (i < NE) { counts[i] = 0; cursors[i] = 0; }
    for (int j = i; j < DIM * DIM; j += stride) Wr16[j] = (_Float16)Wr[j];
    for (int j = i; j < NE * DIM * DIM; j += stride) Wl16[j] = (_Float16)Wl[j];
    for (int j = i; j < DIM * NEP; j += stride) {
        int k = j >> 4, o = j & 15;
        Wd16[j] = (o < NE) ? (_Float16)Wd[k * NE + o] : (_Float16)0.f;
    }
    for (int j = i; j < NE * DIM * NEP; j += stride) {
        int ek = j >> 4, o = j & 15;
        Wld16[j] = (o < NE) ? (_Float16)Wld[ek * NE + o] : (_Float16)0.f;
    }
}

// ---------- K1: h = relu(x @ W_root + b_root), store f16 ----------
// one workgroup = 16 rows x 512 cols; 8 waves, wave w covers cols [w*64, w*64+64)
__global__ __launch_bounds__(256) void k_root(const float* __restrict__ x,
                                              const _Float16* __restrict__ Wr,
                                              const float* __restrict__ b_root,
                                              _Float16* __restrict__ h16) {
    const int wave = threadIdx.x >> 5, lane = threadIdx.x & 31;
    const int khalf = lane >> 4, l16 = lane & 15;
    const int row_base = blockIdx.x * 16;
    const int n_base = wave * 64;
    const float* arow = x + (size_t)(row_base + l16) * DIM + khalf * 8;
    v8f acc[4] = {{0,0,0,0,0,0,0,0},{0,0,0,0,0,0,0,0},{0,0,0,0,0,0,0,0},{0,0,0,0,0,0,0,0}};
    for (int k = 0; k < DIM; k += 32) {
        __builtin_prefetch(arow + k + 32, 0, 1);
        v16h a = load_a_f32(arow + k);
        const _Float16* wrow = Wr + (size_t)(k + khalf * 8) * DIM + n_base + l16;
#pragma unroll
        for (int j = 0; j < 4; ++j) {
            v16h b = load_b(wrow + j * 16, DIM);
            acc[j] = WMMA(acc[j], a, b);
        }
    }
#pragma unroll
    for (int j = 0; j < 4; ++j) {
        const int col = n_base + j * 16 + l16;
        const float bias = b_root[col];
#pragma unroll
        for (int r = 0; r < 8; ++r) {
            int m = r + khalf * 8;
            float v = acc[j][r] + bias;
            v = v > 0.f ? v : 0.f;
            h16[(size_t)(row_base + m) * DIM + col] = (_Float16)v;
        }
    }
}

// ---------- K2: dec = softmax(h @ W_dec), route = argmax; writes out[:,0,:] ----------
// 8 waves per block, each wave owns one 16-row tile, full K=512, Ncols padded to 16
__global__ __launch_bounds__(256) void k_dec(const _Float16* __restrict__ h16,
                                             const _Float16* __restrict__ Wd16,
                                             const float* __restrict__ b_dec,
                                             float* __restrict__ out,
                                             int* __restrict__ route) {
    const int wave = threadIdx.x >> 5, lane = threadIdx.x & 31;
    const int khalf = lane >> 4, l16 = lane & 15;
    const int row_base = (blockIdx.x * 8 + wave) * 16;
    const _Float16* arow = h16 + (size_t)(row_base + l16) * DIM + khalf * 8;
    v8f acc = {0,0,0,0,0,0,0,0};
    for (int k = 0; k < DIM; k += 32) {
        v16h a = load_a_f16(arow + k);
        v16h b = load_b(Wd16 + (size_t)(k + khalf * 8) * NEP + l16, NEP);
        acc = WMMA(acc, a, b);
    }
    const int col = l16;
    const float bias = (col < NE) ? b_dec[col] : 0.f;
#pragma unroll
    for (int r = 0; r < 8; ++r) {
        const int row = row_base + r + khalf * 8;
        float v = acc[r] + bias;
        // reductions stay inside each 8-lane group (cols 0..7 valid)
        float mx = v;
        mx = fmaxf(mx, __shfl_xor(mx, 1, 32));
        mx = fmaxf(mx, __shfl_xor(mx, 2, 32));
        mx = fmaxf(mx, __shfl_xor(mx, 4, 32));
        float ex = __expf(v - mx);
        float s = ex;
        s += __shfl_xor(s, 1, 32); s += __shfl_xor(s, 2, 32); s += __shfl_xor(s, 4, 32);
        float p = ex / s;
        int cand = (v == mx) ? col : (1 << 20);       // first-max tie break like argmax
        cand = min(cand, __shfl_xor(cand, 1, 32));
        cand = min(cand, __shfl_xor(cand, 2, 32));
        cand = min(cand, __shfl_xor(cand, 4, 32));
        if (col < NE) out[(size_t)row * (2 * NE) + col] = p;
        if (col == 0) route[row] = cand;
    }
}

// ---------- K3: expert binning ----------
__global__ void k_count(const int* __restrict__ route, int* __restrict__ counts) {
    int b = blockIdx.x * blockDim.x + threadIdx.x;
    if (b < BN) atomicAdd(&counts[route[b]], 1);
}
__global__ void k_scan_tiles(const int* __restrict__ counts, int* __restrict__ offsets,
                             int* __restrict__ tile_e, int* __restrict__ tile_s,
                             int* __restrict__ tile_end, int* __restrict__ ntiles) {
    if (threadIdx.x == 0 && blockIdx.x == 0) {
        int off = 0, t = 0;
        for (int e = 0; e < NE; ++e) {
            offsets[e] = off;
            int c = counts[e];
            for (int s = 0; s < c; s += 16) {
                tile_e[t] = e;
                tile_s[t] = off + s;
                tile_end[t] = off + ((s + 16 < c) ? (s + 16) : c);
                ++t;
            }
            off += c;
        }
        ntiles[0] = t;
    }
}
__global__ void k_scatter(const int* __restrict__ route, const int* __restrict__ offsets,
                          int* __restrict__ cursors, int* __restrict__ sorted) {
    int b = blockIdx.x * blockDim.x + threadIdx.x;
    if (b < BN) {
        int e = route[b];
        int pos = atomicAdd(&cursors[e], 1);
        sorted[offsets[e] + pos] = b;   // order within expert irrelevant: per-sample results
    }
}

// ---------- K4: grouped leaf GEMM + leaf decision softmax; writes out[:,1,:] ----------
__global__ __launch_bounds__(256) void k_leaf(const _Float16* __restrict__ h16,
                                              const _Float16* __restrict__ Wl16,
                                              const float* __restrict__ b_leaf,
                                              const _Float16* __restrict__ Wld16,
                                              const float* __restrict__ b_ldec,
                                              const int* __restrict__ sorted,
                                              const int* __restrict__ tile_e,
                                              const int* __restrict__ tile_s,
                                              const int* __restrict__ tile_end,
                                              const int* __restrict__ ntiles,
                                              float* __restrict__ out) {
    __shared__ _Float16 lh[16][528];   // 16 rows x 512 cols (+16 pad), 16B-aligned rows
    const int t = blockIdx.x;
    if (t >= ntiles[0]) return;        // uniform across block
    const int e = tile_e[t], start = tile_s[t], endp = tile_end[t];
    const int wave = threadIdx.x >> 5, lane = threadIdx.x & 31;
    const int khalf = lane >> 4, l16 = lane & 15;
    const int slot = start + l16;
    const int ridx = sorted[(slot < endp) ? slot : start];  // clamp padding rows to a valid sample
    const int n_base = wave * 64;
    const _Float16* We = Wl16 + (size_t)e * DIM * DIM;
    const _Float16* arow = h16 + (size_t)ridx * DIM + khalf * 8;
    v8f acc[4] = {{0,0,0,0,0,0,0,0},{0,0,0,0,0,0,0,0},{0,0,0,0,0,0,0,0},{0,0,0,0,0,0,0,0}};
    for (int k = 0; k < DIM; k += 32) {
        __builtin_prefetch(arow + k + 32, 0, 1);
        v16h a = load_a_f16(arow + k);
        const _Float16* wrow = We + (size_t)(k + khalf * 8) * DIM + n_base + l16;
#pragma unroll
        for (int j = 0; j < 4; ++j) {
            v16h b = load_b(wrow + j * 16, DIM);
            acc[j] = WMMA(acc[j], a, b);
        }
    }
#pragma unroll
    for (int j = 0; j < 4; ++j) {
        const int col = n_base + j * 16 + l16;
        const float bias = b_leaf[e * DIM + col];
#pragma unroll
        for (int r = 0; r < 8; ++r) {
            int m = r + khalf * 8;
            float v = acc[j][r] + bias;
            v = v > 0.f ? v : 0.f;
            lh[m][col] = (_Float16)v;
        }
    }
    __syncthreads();
    if (wave == 0) {   // 16x16 decision GEMM from LDS, K=512
        v8f acc2 = {0,0,0,0,0,0,0,0};
        const _Float16* ar = &lh[l16][khalf * 8];
        const _Float16* Wld_e = Wld16 + (size_t)e * DIM * NEP;
        for (int k = 0; k < DIM; k += 32) {
            union { v16h v; v8h h[2]; } u;
            u.h[0] = *(const v8h*)(ar + k);        // ds_load_b128
            u.h[1] = *(const v8h*)(ar + k + 16);
            v16h b = load_b(Wld_e + (size_t)(k + khalf * 8) * NEP + l16, NEP);
            acc2 = WMMA(acc2, u.v, b);
        }
        const int col = l16;
        const float bias = (col < NE) ? b_ldec[e * NE + col] : 0.f;
#pragma unroll
        for (int r = 0; r < 8; ++r) {
            const int m = r + khalf * 8;
            const int slot_m = start + m;
            float v = acc2[r] + bias;
            float mx = v;
            mx = fmaxf(mx, __shfl_xor(mx, 1, 32));
            mx = fmaxf(mx, __shfl_xor(mx, 2, 32));
            mx = fmaxf(mx, __shfl_xor(mx, 4, 32));
            float ex = __expf(v - mx);
            float s = ex;
            s += __shfl_xor(s, 1, 32); s += __shfl_xor(s, 2, 32); s += __shfl_xor(s, 4, 32);
            float p = ex / s;
            if (slot_m < endp && col < NE)
                out[(size_t)sorted[slot_m] * (2 * NE) + NE + col] = p;
        }
    }
}

// ---------- host ----------
extern "C" void kernel_launch(void* const* d_in, const int* in_sizes, int n_in,
                              void* d_out, int out_size, void* d_ws, size_t ws_size,
                              hipStream_t stream) {
    const float* x      = (const float*)d_in[0];
    const float* W_root = (const float*)d_in[1];
    const float* b_root = (const float*)d_in[2];
    const float* W_dec  = (const float*)d_in[3];
    const float* b_dec  = (const float*)d_in[4];
    const float* W_leaf = (const float*)d_in[5];
    const float* b_leaf = (const float*)d_in[6];
    const float* W_ldec = (const float*)d_in[7];
    const float* b_ldec = (const float*)d_in[8];
    float* out = (float*)d_out;

    char* ws = (char*)d_ws;
    size_t off = 0;
    _Float16* h16 = (_Float16*)(ws + off); off += (size_t)BN * DIM * 2;     // 64 MB
    int* route    = (int*)(ws + off);      off += (size_t)BN * 4;
    int* sorted   = (int*)(ws + off);      off += (size_t)BN * 4;
    int* counts   = (int*)(ws + off);      off += 64;
    int* cursors  = (int*)(ws + off);      off += 64;
    int* offsets  = (int*)(ws + off);      off += 64;
    int* ntiles   = (int*)(ws + off);      off += 64;
    int* tile_e   = (int*)(ws + off);      off += (size_t)MAXT * 4;
    int* tile_s   = (int*)(ws + off);      off += (size_t)MAXT * 4;
    int* tile_end = (int*)(ws + off);      off += (size_t)MAXT * 4;
    _Float16* Wr16  = (_Float16*)(ws + off); off += (size_t)DIM * DIM * 2;
    _Float16* Wd16  = (_Float16*)(ws + off); off += (size_t)DIM * NEP * 2;
    _Float16* Wl16  = (_Float16*)(ws + off); off += (size_t)NE * DIM * DIM * 2;
    _Float16* Wld16 = (_Float16*)(ws + off); off += (size_t)NE * DIM * NEP * 2;

    k_prep<<<2048, 256, 0, stream>>>(W_root, W_dec, W_leaf, W_ldec,
                                     Wr16, Wd16, Wl16, Wld16, counts, cursors);
    k_root<<<BN / 16, 256, 0, stream>>>(x, Wr16, b_root, h16);
    k_dec<<<BN / (16 * 8), 256, 0, stream>>>(h16, Wd16, b_dec, out, route);
    k_count<<<BN / 256, 256, 0, stream>>>(route, counts);
    k_scan_tiles<<<1, 32, 0, stream>>>(counts, offsets, tile_e, tile_s, tile_end, ntiles);
    k_scatter<<<BN / 256, 256, 0, stream>>>(route, offsets, cursors, sorted);
    k_leaf<<<BN / 16 + 8, 256, 0, stream>>>(h16, Wl16, b_leaf, Wld16, b_ldec,
                                            sorted, tile_e, tile_s, tile_end, ntiles, out);
}
